// VG_Multihead_Attention_83562883711120
// MI455X (gfx1250) — matrix operand order repaired and checked by reading, not compiled, not run on hardware
//
#include <hip/hip_runtime.h>

// ---------------------------------------------------------------------------
// Problem constants
// ---------------------------------------------------------------------------
#define MD   1024      // model dim
#define KL   4096      // key length
#define NB   16        // batch
#define NH   16        // heads
#define DH   64        // head dim
#define XP   1032      // LDS row pitch (bf16 elems): 1024 + 8 pad -> no bank conflicts
#define MT   64        // rows per block tile
#define THREADS 512    // 16 waves (wave32)

typedef __attribute__((ext_vector_type(16))) __bf16 v16bf;
typedef __attribute__((ext_vector_type(8)))  float  v8f;

union Frag { uint4 q[2]; v16bf v; };

__device__ __forceinline__ unsigned short f2bf(float x) {
    union { float f; unsigned u; } c; c.f = x;
    unsigned r = c.u + 0x7FFFu + ((c.u >> 16) & 1u);   // round-to-nearest-even
    return (unsigned short)(r >> 16);
}

// ---------------------------------------------------------------------------
// Kernel 0: fp32 -> bf16 conversion of Wv and Wo (weights stay hot in L2)
// ---------------------------------------------------------------------------
__global__ void k_cvt(const float* __restrict__ Wv, const float* __restrict__ Wo,
                      unsigned short* __restrict__ Wv_bf, unsigned short* __restrict__ Wo_bf) {
    int i = blockIdx.x * 256 + threadIdx.x;           // 2M threads
    if (i < MD * MD) Wv_bf[i] = f2bf(Wv[i]);
    else             Wo_bf[i - MD * MD] = f2bf(Wo[i - MD * MD]);
}

// ---------------------------------------------------------------------------
// Kernel 1: q_s[b, j] = (hidden[b,:] . Wq[j,:] + bq[j]) / sqrt(DH)
// ---------------------------------------------------------------------------
__global__ void k_qproj(const float* __restrict__ hid, const float* __restrict__ Wq,
                        const float* __restrict__ bq, float* __restrict__ q_s) {
    int i = blockIdx.x * 256 + threadIdx.x;           // NB*MD = 16384 threads
    int b = i >> 10, j = i & 1023;
    const float4* h4 = (const float4*)(hid + (size_t)b * MD);
    const float4* w4 = (const float4*)(Wq + (size_t)j * MD);
    float acc = 0.f;
    for (int m = 0; m < MD / 4; ++m) {
        float4 a = h4[m], w = w4[m];
        acc += a.x * w.x + a.y * w.y + a.z * w.z + a.w * w.w;
    }
    q_s[i] = (acc + bq[j]) * 0.125f;                  // 1/sqrt(64)
}

// ---------------------------------------------------------------------------
// Kernel 2: fold Wk into per-(b,head) vectors: u[b,h,m] = sum_d q_s[b,h,d]*Wk[h*64+d, m]
// stored bf16, row-major [b*16+h][1024]  (serves as WMMA B-matrix rows)
// ---------------------------------------------------------------------------
__global__ void k_uproj(const float* __restrict__ Wk, const float* __restrict__ q_s,
                        unsigned short* __restrict__ U_bf) {
    int i = blockIdx.x * 256 + threadIdx.x;           // NB*NH*MD = 262144 threads
    int b = i >> 14, rem = i & 16383;
    int h = rem >> 10, m = rem & 1023;
    const float* qs = q_s + b * MD + h * DH;
    const float* wk = Wk + (size_t)(h * DH) * MD + m;
    float acc = 0.f;
    for (int d = 0; d < DH; ++d) acc += wk[(size_t)d * MD] * qs[d];
    U_bf[((size_t)(b * NH + h) << 10) + m] = f2bf(acc);
}

// ---------------------------------------------------------------------------
// Kernel 3: score bias sb[b,h] = sum_d bk[h*64+d]*q_s[b,h,d]
// ---------------------------------------------------------------------------
__global__ void k_sb(const float* __restrict__ bk, const float* __restrict__ q_s,
                     float* __restrict__ sb) {
    int i = threadIdx.x;                              // 256 threads, 1 block
    int b = i >> 4, h = i & 15;
    float acc = 0.f;
    for (int d = 0; d < DH; ++d) acc += bk[h * DH + d] * q_s[b * MD + h * DH + d];
    sb[i] = acc;
}

// ---------------------------------------------------------------------------
// Main fused kernel: per 64-row tile of one batch
//   Phase 0: kv tile fp32 -> bf16 in LDS
//   Phase 1: attn = clip(X @ U_b^T + sb, 0, 1)        (WMMA, N = 16 heads)
//   Phase 2: ctx  = attn ⊙ (X @ Wv^T + bv)  -> LDS    (WMMA)
//   Phase 3: out  = ctx @ Wo^T + bo         -> HBM    (WMMA)
//
// Wave->tile mapping for phases 2/3: wave w owns ALL 4 M-subtiles and the
// 4 N-subtiles of columns [w*64, w*64+64) — i.e. exactly one head. B-tiles
// are thus partitioned across waves with no intra-block duplication:
// per chunk a wave loads 4 B-tiles (4 KB) and issues 16 WMMAs.
// ---------------------------------------------------------------------------
__global__ __launch_bounds__(THREADS)
void k_main(const float* __restrict__ kv,
            const unsigned short* __restrict__ Wv_bf,
            const unsigned short* __restrict__ Wo_bf,
            const unsigned short* __restrict__ U_bf,
            const float* __restrict__ sb,
            const float* __restrict__ bv,
            const float* __restrict__ bo,
            float* __restrict__ out) {
    extern __shared__ unsigned short smem[];
    unsigned short* Xs = smem;                         // 64 x 1032 bf16
    unsigned short* Cs = smem + MT * XP;               // 64 x 1032 bf16
    float* attnS = (float*)(smem + 2 * MT * XP);       // 64 x 16 fp32

    const int tid  = threadIdx.x;
    const int lane = tid & 31;
    const int ln   = lane & 15;
    const int hi   = lane >> 4;                        // 0 | 1 (upper half-wave)
    const int w    = tid >> 5;                         // wave id 0..15
    const int b    = blockIdx.x >> 6;                  // batch
    const int row0 = (blockIdx.x & 63) << 6;           // key row base within batch
    const long g0  = (long)b * KL + row0;              // global row base

    const v8f vzero = {};

    // ---------------- Phase 0: load/convert X tile ----------------
    for (int i = tid; i < MT * (MD / 4); i += THREADS) {
        int r = i >> 8;
        int c = (i & 255) << 2;
        float4 f = *(const float4*)(kv + (g0 + r) * (long)MD + c);
        unsigned short* d = Xs + r * XP + c;
        d[0] = f2bf(f.x); d[1] = f2bf(f.y); d[2] = f2bf(f.z); d[3] = f2bf(f.w);
    }
    __syncthreads();

    // ---------------- Phase 1: attention gate (waves 0..3) ----------------
    if (w < 4) {
        const int m0 = w << 4;
        v8f acc = {};
        const unsigned short* Ub = U_bf + (size_t)((b << 4) + ln) * MD;
        const unsigned short* Ar = Xs + (m0 + ln) * XP + (hi << 4);
        for (int kc = 0; kc < 32; ++kc) {
            const int c0 = kc << 5;
            Frag a, bb;
            a.q[0] = *(const uint4*)(Ar + c0);
            a.q[1] = *(const uint4*)(Ar + c0 + 8);
            const uint4* bp = (const uint4*)(Ub + c0 + (hi << 4));
            bb.q[0] = bp[0]; bb.q[1] = bp[1];
            acc = __builtin_amdgcn_wmma_f32_16x16x32_bf16(
                false, a.v, false, bb.v, (short)0, acc, false, false);
        }
        const float sbv = sb[(b << 4) + ln];
        #pragma unroll
        for (int r = 0; r < 8; ++r) {
            float s = acc[r] + sbv;
            s = fminf(fmaxf(s, 0.0f), 1.0f);           // Hardtanh(0,1)
            attnS[(m0 + r + (hi << 3)) * 16 + ln] = s;
        }
    }
    __syncthreads();

    // ---------------- Phase 2: ctx = attn * (X @ Wv^T + bv) ----------------
    {
        const int nbase = w << 6;                      // this wave's 64-col span = head w
        v8f acc[4][4];
        #pragma unroll
        for (int ms = 0; ms < 4; ++ms)
            #pragma unroll
            for (int ns = 0; ns < 4; ++ns) acc[ms][ns] = vzero;

        for (int kc = 0; kc < 32; ++kc) {
            const int c0 = kc << 5;
            Frag bfr[4];
            #pragma unroll
            for (int ns = 0; ns < 4; ++ns) {
                const int n = nbase + (ns << 4) + ln;
                const uint4* bp = (const uint4*)(Wv_bf + (size_t)n * MD + c0 + (hi << 4));
                bfr[ns].q[0] = bp[0]; bfr[ns].q[1] = bp[1];
            }
            #pragma unroll
            for (int ms = 0; ms < 4; ++ms) {
                Frag a;
                const unsigned short* Ar = Xs + ((ms << 4) + ln) * XP + c0 + (hi << 4);
                a.q[0] = *(const uint4*)(Ar);
                a.q[1] = *(const uint4*)(Ar + 8);
                #pragma unroll
                for (int ns = 0; ns < 4; ++ns)
                    acc[ms][ns] = __builtin_amdgcn_wmma_f32_16x16x32_bf16(
                        false, a.v, false, bfr[ns].v, (short)0, acc[ms][ns], false, false);
            }
        }
        // epilogue: add bias, apply gate (head == w for this wave), store bf16 ctx
        #pragma unroll
        for (int ms = 0; ms < 4; ++ms) {
            #pragma unroll
            for (int ns = 0; ns < 4; ++ns) {
                const int n = nbase + (ns << 4) + ln;
                const float bvn = bv[n];
                #pragma unroll
                for (int r = 0; r < 8; ++r) {
                    const int m = (ms << 4) + r + (hi << 3);
                    float vv = (acc[ms][ns][r] + bvn) * attnS[m * 16 + w];
                    Cs[m * XP + n] = f2bf(vv);
                }
            }
        }
    }
    __syncthreads();

    // ---------------- Phase 3: out = ctx @ Wo^T + bo ----------------
    {
        const int nbase = w << 6;
        v8f acc[4][4];
        #pragma unroll
        for (int ms = 0; ms < 4; ++ms)
            #pragma unroll
            for (int ns = 0; ns < 4; ++ns) acc[ms][ns] = vzero;

        for (int kc = 0; kc < 32; ++kc) {
            const int c0 = kc << 5;
            Frag bfr[4];
            #pragma unroll
            for (int ns = 0; ns < 4; ++ns) {
                const int n = nbase + (ns << 4) + ln;
                const uint4* bp = (const uint4*)(Wo_bf + (size_t)n * MD + c0 + (hi << 4));
                bfr[ns].q[0] = bp[0]; bfr[ns].q[1] = bp[1];
            }
            #pragma unroll
            for (int ms = 0; ms < 4; ++ms) {
                Frag a;
                const unsigned short* Ar = Cs + ((ms << 4) + ln) * XP + c0 + (hi << 4);
                a.q[0] = *(const uint4*)(Ar);
                a.q[1] = *(const uint4*)(Ar + 8);
                #pragma unroll
                for (int ns = 0; ns < 4; ++ns)
                    acc[ms][ns] = __builtin_amdgcn_wmma_f32_16x16x32_bf16(
                        false, a.v, false, bfr[ns].v, (short)0, acc[ms][ns], false, false);
            }
        }
        #pragma unroll
        for (int ms = 0; ms < 4; ++ms) {
            #pragma unroll
            for (int ns = 0; ns < 4; ++ns) {
                const int n = nbase + (ns << 4) + ln;
                const float bon = bo[n];
                #pragma unroll
                for (int r = 0; r < 8; ++r) {
                    const int m = (ms << 4) + r + (hi << 3);
                    out[(g0 + m) * (long)MD + n] = acc[ms][ns][r] + bon;
                }
            }
        }
    }
}

// ---------------------------------------------------------------------------
// Launcher
// ---------------------------------------------------------------------------
extern "C" void kernel_launch(void* const* d_in, const int* in_sizes, int n_in,
                              void* d_out, int out_size, void* d_ws, size_t ws_size,
                              hipStream_t stream) {
    const float* hid = (const float*)d_in[0];   // [16, 1, 1024]
    const float* kv  = (const float*)d_in[1];   // [16, 4096, 1024]
    const float* Wq  = (const float*)d_in[2];
    const float* bq  = (const float*)d_in[3];
    const float* Wk  = (const float*)d_in[4];
    const float* bk  = (const float*)d_in[5];
    const float* Wv  = (const float*)d_in[6];
    const float* bv  = (const float*)d_in[7];
    const float* Wo  = (const float*)d_in[8];
    const float* bo  = (const float*)d_in[9];
    float* out = (float*)d_out;

    // workspace carve-out (~4.8 MB total)
    unsigned short* Wv_bf = (unsigned short*)d_ws;                  // 2 MB
    unsigned short* Wo_bf = Wv_bf + (size_t)MD * MD;                // 2 MB
    unsigned short* U_bf  = Wo_bf + (size_t)MD * MD;                // 512 KB
    float* q_s = (float*)(U_bf + (size_t)NB * NH * MD);             // 64 KB
    float* sbw = q_s + (size_t)NB * MD;                             // 1 KB

    // weight conversion (2M elems)
    k_cvt<<<(2 * MD * MD) / 256, 256, 0, stream>>>(Wv, Wo, Wv_bf, Wo_bf);
    // q projection (scaled)
    k_qproj<<<(NB * MD) / 256, 256, 0, stream>>>(hid, Wq, bq, q_s);
    // fold Wk through q -> per-head score vectors
    k_uproj<<<(NB * NH * MD) / 256, 256, 0, stream>>>(Wk, q_s, U_bf);
    k_sb<<<1, NB * NH, 0, stream>>>(bk, q_s, sbw);

    // fused attention + V-proj + O-proj
    const size_t lds_bytes = (size_t)2 * MT * XP * sizeof(unsigned short)
                           + (size_t)MT * NH * sizeof(float);       // 268,288 B
    (void)hipFuncSetAttribute((const void*)k_main,
                              hipFuncAttributeMaxDynamicSharedMemorySize,
                              (int)lds_bytes);
    k_main<<<(NB * KL) / MT, THREADS, lds_bytes, stream>>>(
        kv, Wv_bf, Wo_bf, U_bf, sbw, bv, bo, out);
}
